// DFlashModel_50525995270366
// MI455X (gfx1250) — matrix operand order repaired and checked by reading, not compiled, not run on hardware
//
#include <hip/hip_runtime.h>
#include <hip/hip_bf16.h>

#define DEV __device__ __forceinline__

// ---- problem constants ----
#define LL 3
#define BB 2
#define SS 2048
#define DD 1024
#define HH 8
#define DHH 128
#define NAA 128
#define BSS 16
#define QQ 2048           // NA*BS
#define KVV 4096          // S + Q
#define VV 32000
#define MASK_ID (VV - 1)
#define NEGF (-3.0e38f)

#define AS1 __attribute__((address_space(1)))
#define AS3 __attribute__((address_space(3)))

#if __has_builtin(__builtin_amdgcn_global_load_async_to_lds_b128)
#define HAVE_ASYNC_LDS 1
#else
#define HAVE_ASYNC_LDS 0
#endif

// ---- WMMA types (gfx1250, wave32) ----
typedef __bf16 bf16x16 __attribute__((ext_vector_type(16)));
typedef float  f32x8   __attribute__((ext_vector_type(8)));
typedef int    v4i     __attribute__((ext_vector_type(4)));

union FragU { uint4 q[2]; bf16x16 v; };

DEV unsigned short f2bf(float f) {
  unsigned int u = __float_as_uint(f);
  unsigned int r = u + 0x7FFFu + ((u >> 16) & 1u);
  return (unsigned short)(r >> 16);
}

// A fragment: A row-major [M,K]; lane l<16 -> row m0+l, K = {8hi..8hi+7, 16+8hi..16+8hi+7}
DEV bf16x16 load_frag_a(const unsigned short* A, int lda, int m0, int k0, int lane) {
  int row = lane & 15, hi = lane >> 4;
  const unsigned short* p = A + (size_t)(m0 + row) * lda + k0 + 8 * hi;
  FragU u;
  u.q[0] = *(const uint4*)(p);
  u.q[1] = *(const uint4*)(p + 16);
  return u.v;
}

// B fragment: B row-major [N,K] (computing A*B^T); lane -> col n0+(l&15), K = {8hi..8hi+7, 16+8hi..16+8hi+7}
DEV bf16x16 load_frag_b(const unsigned short* B, int ldb, int n0, int k0, int lane) {
  int col = lane & 15, hi = lane >> 4;
  const unsigned short* p = B + (size_t)(n0 + col) * ldb + k0 + 8 * hi;
  FragU u;
  u.q[0] = *(const uint4*)(p);
  u.q[1] = *(const uint4*)(p + 16);
  return u.v;
}

DEV f32x8 wmma_bf16(bf16x16 a, bf16x16 b, f32x8 c) {
  return __builtin_amdgcn_wmma_f32_16x16x32_bf16(false, a, false, b, (short)0, c, false, false);
}

// ====================== elementwise / prep kernels ======================

__global__ void k_zero4(float* a) { if (threadIdx.x < 4) a[threadIdx.x] = 0.f; }

__global__ void k_cvt_bf16(const float* __restrict__ in, unsigned short* __restrict__ out, long n) {
  long i = (long)blockIdx.x * blockDim.x + threadIdx.x;
  long st = (long)gridDim.x * blockDim.x;
  for (; i < n; i += st) out[i] = f2bf(in[i]);
}

// hcat[b,s, l*D + d] = hidden[l,b,s,d]   (written directly as bf16)
__global__ void k_hcat(const float* __restrict__ hs, unsigned short* __restrict__ out, long n) {
  long i = (long)blockIdx.x * blockDim.x + threadIdx.x;
  long st = (long)gridDim.x * blockDim.x;
  for (; i < n; i += st) {
    long d = i & (DD - 1);
    long t = i >> 10;
    int  l = (int)(t % LL);
    long t2 = t / LL;
    long s = t2 & (SS - 1);
    long b = t2 >> 11;
    out[i] = f2bf(hs[((((long)l * BB + b) * SS + s) << 10) + d]);
  }
}

__global__ void k_build_ids(const int* __restrict__ input_ids, const int* __restrict__ anchors,
                            int* __restrict__ draft, int* __restrict__ lbls, int* __restrict__ posf) {
  int idx = blockIdx.x * blockDim.x + threadIdx.x;
  if (idx >= BB * QQ) return;
  int qi = idx & (QQ - 1);
  int b  = idx >> 11;
  int pos = anchors[b * NAA + (qi >> 4)] + (qi & 15);
  bool inb = pos < SS;
  int g = inb ? pos : SS - 1;
  int tok = input_ids[b * SS + g];
  draft[idx] = ((qi & 15) == 0) ? tok : MASK_ID;
  lbls[idx]  = (((qi & 15) != 0) && inb) ? tok : -100;
  posf[idx]  = pos;
}

__global__ void k_emb_gather(const float* __restrict__ table, const int* __restrict__ draft,
                             unsigned short* __restrict__ eb, float* __restrict__ ef, long n) {
  long i = (long)blockIdx.x * blockDim.x + threadIdx.x;
  long st = (long)gridDim.x * blockDim.x;
  for (; i < n; i += st) {
    long d = i & (DD - 1);
    long row = i >> 10;
    int id = draft[row];
    float v = table[(size_t)id * DD + d];
    ef[i] = v;
    eb[i] = f2bf(v);
  }
}

// RoPE on fp32 input -> bf16 output.  rows = B*Q (is_k=0) or B*KV (is_k=1)
__global__ void k_rope(const float* __restrict__ x, const int* __restrict__ posf,
                       unsigned short* __restrict__ outB, long n, int is_k) {
  long i = (long)blockIdx.x * blockDim.x + threadIdx.x;
  long st = (long)gridDim.x * blockDim.x;
  for (; i < n; i += st) {
    int d = (int)(i & 63);
    long t = i >> 6;
    int hh = (int)(t & (HH - 1));
    long row = t >> 3;
    int pos;
    if (is_k) {
      int r = (int)(row & (KVV - 1));
      int b = (int)(row >> 12);
      pos = (r < SS) ? r : posf[b * QQ + (r - SS)];
    } else {
      pos = posf[row];
    }
    float inv = __expf(-(float)d * (9.210340371976184f / 64.f));  // 10000^(-d/64)
    float ang = (float)pos * inv;
    float sn, cs;
    __sincosf(ang, &sn, &cs);
    size_t base = (size_t)row * DD + hh * DHH + d;
    float x1 = x[base], x2 = x[base + 64];
    outB[base]      = f2bf(x1 * cs - x2 * sn);
    outB[base + 64] = f2bf(x1 * sn + x2 * cs);
  }
}

// vT[b][h][d][kv] = v[b][kv][h*DH + d]  (fp32 -> bf16)
__global__ void k_vtrans(const float* __restrict__ vf, unsigned short* __restrict__ vT, long n) {
  long i = (long)blockIdx.x * blockDim.x + threadIdx.x;
  long st = (long)gridDim.x * blockDim.x;
  for (; i < n; i += st) {
    int kv = (int)(i & (KVV - 1));
    long t = i >> 12;
    int d = (int)(t & (DHH - 1));
    long t2 = t >> 7;
    int h = (int)(t2 & (HH - 1));
    int b = (int)(t2 >> 3);
    vT[i] = f2bf(vf[((size_t)(b * KVV + kv)) * DD + h * DHH + d]);
  }
}

// ====================== generic WMMA GEMM: C[M,N] = A[M,K] * B[N,K]^T ======================
// 8 waves (4x2), each wave owns a 32x32 register tile (2x2 WMMA tiles) -> 128x64 block tile.
// K-loop unrolled by 2 (K % 64 == 0): 16 clause-grouped b128 loads feed 8 back-to-back WMMAs,
// with no cross-iteration register rotation (avoids v_mov chains).
__global__ void __launch_bounds__(256)
k_gemm(const unsigned short* __restrict__ A, const unsigned short* __restrict__ B,
       void* __restrict__ Cv, int M, int N, int K, int lda, int ldb, int ldc, int out_bf16) {
  int lane = threadIdx.x & 31;
  int wave = threadIdx.x >> 5;
  int m0 = blockIdx.y * 128 + (wave >> 1) * 32;
  int n0 = blockIdx.x * 64 + (wave & 1) * 32;
  if (m0 >= M || n0 >= N) return;
  int row = lane & 15, hi = lane >> 4, col = lane & 15;
  const f32x8 zed = {0.f, 0.f, 0.f, 0.f, 0.f, 0.f, 0.f, 0.f};
  f32x8 acc00 = zed, acc01 = zed, acc10 = zed, acc11 = zed;

  for (int k0 = 0; k0 < K; k0 += 64) {
    bf16x16 a0 = load_frag_a(A, lda, m0, k0, lane);
    bf16x16 a1 = load_frag_a(A, lda, m0 + 16, k0, lane);
    bf16x16 b0 = load_frag_b(B, ldb, n0, k0, lane);
    bf16x16 b1 = load_frag_b(B, ldb, n0 + 16, k0, lane);
    bf16x16 a2 = load_frag_a(A, lda, m0, k0 + 32, lane);
    bf16x16 a3 = load_frag_a(A, lda, m0 + 16, k0 + 32, lane);
    bf16x16 b2 = load_frag_b(B, ldb, n0, k0 + 32, lane);
    bf16x16 b3 = load_frag_b(B, ldb, n0 + 16, k0 + 32, lane);
    if (k0 + 64 < K) {  // keep next slab flowing into caches
      __builtin_prefetch(A + (size_t)(m0 + row) * lda + k0 + 64, 0, 1);
      __builtin_prefetch(A + (size_t)(m0 + 16 + row) * lda + k0 + 64, 0, 1);
      __builtin_prefetch(B + (size_t)(n0 + col) * ldb + k0 + 64, 0, 1);
      __builtin_prefetch(B + (size_t)(n0 + 16 + col) * ldb + k0 + 64, 0, 1);
    }
    acc00 = wmma_bf16(a0, b0, acc00);
    acc01 = wmma_bf16(a0, b1, acc01);
    acc10 = wmma_bf16(a1, b0, acc10);
    acc11 = wmma_bf16(a1, b1, acc11);
    acc00 = wmma_bf16(a2, b2, acc00);
    acc01 = wmma_bf16(a2, b3, acc01);
    acc10 = wmma_bf16(a3, b2, acc10);
    acc11 = wmma_bf16(a3, b3, acc11);
  }
  if (out_bf16) {
    unsigned short* C = (unsigned short*)Cv;
    for (int j = 0; j < 8; ++j) {
      C[(size_t)(m0 + j + 8 * hi) * ldc + n0 + col]           = f2bf(acc00[j]);
      C[(size_t)(m0 + j + 8 * hi) * ldc + n0 + 16 + col]      = f2bf(acc01[j]);
      C[(size_t)(m0 + 16 + j + 8 * hi) * ldc + n0 + col]      = f2bf(acc10[j]);
      C[(size_t)(m0 + 16 + j + 8 * hi) * ldc + n0 + 16 + col] = f2bf(acc11[j]);
    }
  } else {
    float* C = (float*)Cv;
    for (int j = 0; j < 8; ++j) {
      C[(size_t)(m0 + j + 8 * hi) * ldc + n0 + col]           = acc00[j];
      C[(size_t)(m0 + j + 8 * hi) * ldc + n0 + 16 + col]      = acc01[j];
      C[(size_t)(m0 + 16 + j + 8 * hi) * ldc + n0 + col]      = acc10[j];
      C[(size_t)(m0 + 16 + j + 8 * hi) * ldc + n0 + 16 + col] = acc11[j];
    }
  }
}

// ====================== flash attention (anchor-masked) ======================
// grid: (Q/128, H, B); each wave owns a 16-row Q tile, streams KV in 32-wide tiles.
__global__ void __launch_bounds__(256)
k_attn(const unsigned short* __restrict__ qb, const unsigned short* __restrict__ kb,
       const unsigned short* __restrict__ vT, const int* __restrict__ anchors,
       float* __restrict__ out) {
  __shared__ unsigned short Pl[8][16 * 32];
  int lane = threadIdx.x & 31;
  int wave = threadIdx.x >> 5;
  int b = blockIdx.z, h = blockIdx.y;
  int q0 = blockIdx.x * 128 + wave * 16;   // q row base within batch
  int hi = lane >> 4, col = lane & 15;

  const unsigned short* Qbase = qb + (size_t)b * QQ * DD + h * DHH;
  const unsigned short* Kbase = kb + (size_t)b * KVV * DD + h * DHH;
  const unsigned short* Vt    = vT + ((size_t)(b * HH + h)) * DHH * KVV;

  int anc = anchors[b * NAA + (q0 >> 4)];
  int blk_lo = SS + (q0 >> 4) * BSS;
  int blk_hi = blk_lo + BSS;

  // Q-tile A fragments are loop invariant: hoist all 4 (K = 0,32,64,96)
  bf16x16 aq[4];
  for (int k = 0; k < 4; ++k) aq[k] = load_frag_a(Qbase, DD, q0, k * 32, lane);

  float mrun[8], lrun[8];
  f32x8 o[8];
  const f32x8 zed = {0.f, 0.f, 0.f, 0.f, 0.f, 0.f, 0.f, 0.f};
  for (int j = 0; j < 8; ++j) { mrun[j] = NEGF; lrun[j] = 0.f; o[j] = zed; }

  const float sc = 0.08838834764831845f;  // 1/sqrt(128)
  unsigned short* myP = &Pl[wave][0];

  for (int kv0 = 0; kv0 < KVV; kv0 += 32) {
    f32x8 s0 = zed, s1 = zed;
    for (int k = 0; k < 4; ++k) {
      s0 = wmma_bf16(aq[k], load_frag_b(Kbase, DD, kv0, k * 32, lane), s0);
      s1 = wmma_bf16(aq[k], load_frag_b(Kbase, DD, kv0 + 16, k * 32, lane), s1);
    }
    int c0 = kv0 + col, c1 = kv0 + 16 + col;
    bool v0 = (c0 < anc) || (c0 >= blk_lo && c0 < blk_hi);
    bool v1 = (c1 < anc) || (c1 >= blk_lo && c1 < blk_hi);
    float resc[8];
    for (int j = 0; j < 8; ++j) {
      float x0 = v0 ? s0[j] * sc : -1e30f;
      float x1 = v1 ? s1[j] * sc : -1e30f;
      float mx = fmaxf(x0, x1);
      for (int off = 8; off; off >>= 1) mx = fmaxf(mx, __shfl_xor(mx, off, 32));
      float mn = fmaxf(mrun[j], mx);
      float p0 = __expf(x0 - mn), p1 = __expf(x1 - mn);
      float ps = p0 + p1;
      for (int off = 8; off; off >>= 1) ps += __shfl_xor(ps, off, 32);
      resc[j] = __expf(mrun[j] - mn);
      lrun[j] = lrun[j] * resc[j] + ps;
      mrun[j] = mn;
      myP[(j + 8 * hi) * 32 + col]      = f2bf(p0);
      myP[(j + 8 * hi) * 32 + 16 + col] = f2bf(p1);
    }
    asm volatile("s_wait_dscnt 0x0" ::: "memory");
    bf16x16 ap = load_frag_a(myP, 32, 0, 0, lane);
    for (int dt = 0; dt < 8; ++dt) {
      bf16x16 bv = load_frag_b(Vt, KVV, dt * 16, kv0, lane);
      f32x8 t = o[dt];
      for (int j = 0; j < 8; ++j) t[j] *= resc[j];
      o[dt] = wmma_bf16(ap, bv, t);
    }
  }
  for (int j = 0; j < 8; ++j) {
    float inv = 1.f / lrun[j];
    int r = q0 + j + 8 * hi;
    for (int dt = 0; dt < 8; ++dt)
      out[((size_t)b * QQ + r) * DD + h * DHH + dt * 16 + col] = o[dt][j] * inv;
  }
}

// ====================== residual + RMSNorm -> bf16 ======================
__global__ void __launch_bounds__(256)
k_rmsnorm(const float* __restrict__ emb, const float* __restrict__ proj,
          const float* __restrict__ g, unsigned short* __restrict__ outB) {
  __shared__ float red[8];
  int row = blockIdx.x;
  float hv[4];
  float local = 0.f;
  for (int i = 0; i < 4; ++i) {
    int d = threadIdx.x + i * 256;
    float x = emb[(size_t)row * DD + d] + proj[(size_t)row * DD + d];
    hv[i] = x;
    local += x * x;
  }
  for (int off = 16; off; off >>= 1) local += __shfl_xor(local, off, 32);
  if ((threadIdx.x & 31) == 0) red[threadIdx.x >> 5] = local;
  __syncthreads();
  float sum = 0.f;
  for (int w = 0; w < 8; ++w) sum += red[w];
  float scale = rsqrtf(sum / (float)DD + 1e-6f);
  for (int i = 0; i < 4; ++i) {
    int d = threadIdx.x + i * 256;
    outB[(size_t)row * DD + d] = f2bf(hv[i] * scale * g[d]);
  }
}

// ====================== lm_head GEMM + streaming log-softmax / NLL / argmax ======================
// grid: B*Q/16 blocks. The 16-row h tile (32 KB bf16) is staged into LDS once via the async
// tensor path, then 8 waves sweep the 1000 vocab tile-pairs with streaming stats.
__global__ void __launch_bounds__(256)
k_lmloss(const unsigned short* __restrict__ hB, const unsigned short* __restrict__ W,
         const int* __restrict__ labels, float* __restrict__ accum) {
  __shared__ unsigned short hTile[16 * DD];
  __shared__ float Lm[8][16], Ls[8][16], Lav[8][16], Llb[8][16];
  __shared__ int   Lai[8][16];
  int lane = threadIdx.x & 31, wave = threadIdx.x >> 5;
  int hi = lane >> 4, col = lane & 15;
  int m0 = blockIdx.x * 16;

  // stage the h tile into LDS (16 rows x 1024 cols bf16, 2048 x 16B chunks)
  for (int c = threadIdx.x; c < 16 * 128; c += 256) {
    int r = c >> 7, off = (c & 127) * 8;
    const unsigned short* g = hB + (size_t)(m0 + r) * DD + off;
#if HAVE_ASYNC_LDS
    __builtin_amdgcn_global_load_async_to_lds_b128(
        (AS1 v4i*)g, (AS3 v4i*)&hTile[r * DD + off], 0, 0);
#else
    *(uint4*)&hTile[r * DD + off] = *(const uint4*)g;
#endif
  }
#if HAVE_ASYNC_LDS
  asm volatile("s_wait_asynccnt 0x0" ::: "memory");
#endif
  __syncthreads();

  float m[8], s[8], av[8], lb[8];
  int ai[8], lblc[8];
  for (int j = 0; j < 8; ++j) {
    m[j] = NEGF; s[j] = 0.f; av[j] = NEGF; lb[j] = NEGF; ai[j] = 0;
    int l_ = labels[m0 + j + 8 * hi];
    lblc[j] = l_ < 0 ? 0 : l_;
  }
  // two vocab tiles per iteration, A fragment (from LDS) reused by both WMMAs
  for (int pt = wave; pt < VV / 32; pt += 8) {
    int n0 = pt * 32;
    f32x8 acc0 = {0.f, 0.f, 0.f, 0.f, 0.f, 0.f, 0.f, 0.f};
    f32x8 acc1 = acc0;
    for (int k0 = 0; k0 < DD; k0 += 32) {
      if (k0 + 64 < DD) {
        __builtin_prefetch(W + (size_t)(n0 + col) * DD + k0 + 64, 0, 1);
        __builtin_prefetch(W + (size_t)(n0 + 16 + col) * DD + k0 + 64, 0, 1);
      }
      bf16x16 a = load_frag_a(hTile, DD, 0, k0, lane);
      acc0 = wmma_bf16(a, load_frag_b(W, DD, n0, k0, lane), acc0);
      acc1 = wmma_bf16(a, load_frag_b(W, DD, n0 + 16, k0, lane), acc1);
    }
    int c0 = n0 + col, c1 = n0 + 16 + col;
    for (int j = 0; j < 8; ++j) {
      float v0 = acc0[j], v1 = acc1[j];
      float mx = fmaxf(v0, v1);
      for (int off = 8; off; off >>= 1) mx = fmaxf(mx, __shfl_xor(mx, off, 32));
      float mn = fmaxf(m[j], mx);
      float p = __expf(v0 - mn) + __expf(v1 - mn);
      for (int off = 8; off; off >>= 1) p += __shfl_xor(p, off, 32);
      s[j] = s[j] * __expf(m[j] - mn) + p;
      m[j] = mn;
      if (v0 > av[j]) { av[j] = v0; ai[j] = c0; }
      if (v1 > av[j]) { av[j] = v1; ai[j] = c1; }
      if (c0 == lblc[j]) lb[j] = v0;
      if (c1 == lblc[j]) lb[j] = v1;
    }
  }
  // cross-lane merge within each 16-lane half
  for (int j = 0; j < 8; ++j) {
    float mm = m[j], ss = s[j], vv = av[j], ll = lb[j];
    int ii = ai[j];
    for (int off = 8; off; off >>= 1) {
      float mo = __shfl_xor(mm, off, 32), so = __shfl_xor(ss, off, 32);
      float M = fmaxf(mm, mo);
      ss = ss * __expf(mm - M) + so * __expf(mo - M);
      mm = M;
      float vo = __shfl_xor(vv, off, 32);
      int io = __shfl_xor(ii, off, 32);
      if (vo > vv || (vo == vv && io < ii)) { vv = vo; ii = io; }
      ll = fmaxf(ll, __shfl_xor(ll, off, 32));
    }
    if (col == 0) {
      int r = j + 8 * hi;
      Lm[wave][r] = mm; Ls[wave][r] = ss; Lav[wave][r] = vv; Lai[wave][r] = ii; Llb[wave][r] = ll;
    }
  }
  __syncthreads();
  if (threadIdx.x < 16) {
    int r = threadIdx.x;
    float mm = Lm[0][r], ss = Ls[0][r], vv = Lav[0][r], ll = Llb[0][r];
    int ii = Lai[0][r];
    for (int w = 1; w < 8; ++w) {
      float mo = Lm[w][r], so = Ls[w][r];
      float M = fmaxf(mm, mo);
      ss = ss * __expf(mm - M) + so * __expf(mo - M);
      mm = M;
      float vo = Lav[w][r];
      int io = Lai[w][r];
      if (vo > vv || (vo == vv && io < ii)) { vv = vo; ii = io; }
      ll = fmaxf(ll, Llb[w][r]);
    }
    float nll = mm + __logf(ss) - ll;
    int lbl = labels[m0 + r];
    bool valid = lbl != -100;
    int p = (m0 + r) & 15;
    float decay = (p == 0) ? 0.f : __expf(-((float)p - 1.f) / 7.f);
    float wt = valid ? decay : 0.f;
    atomicAdd(accum + 0, wt * nll);
    atomicAdd(accum + 1, wt);
    atomicAdd(accum + 2, (valid && ii == lbl) ? 1.f : 0.f);
    atomicAdd(accum + 3, valid ? 1.f : 0.f);
  }
}

__global__ void k_final(const float* __restrict__ a, float* __restrict__ out) {
  if (threadIdx.x == 0) {
    out[0] = a[0] / fmaxf(a[1], 1e-6f);
    out[1] = a[2] / fmaxf(a[3], 1.f);
  }
}

// ====================== host side ======================
extern "C" void kernel_launch(void* const* d_in, const int* in_sizes, int n_in,
                              void* d_out, int out_size, void* d_ws, size_t ws_size,
                              hipStream_t stream) {
  const int*   input_ids = (const int*)d_in[0];
  const float* hidden    = (const float*)d_in[1];
  const int*   anchors   = (const int*)d_in[2];
  const float* lm_w      = (const float*)d_in[3];
  const float* norm_w    = (const float*)d_in[4];
  const float* fc_w      = (const float*)d_in[5];
  const float* embed     = (const float*)d_in[6];
  const float* wq        = (const float*)d_in[7];
  const float* wk        = (const float*)d_in[8];
  const float* wv        = (const float*)d_in[9];
  const float* wo        = (const float*)d_in[10];
  float* out = (float*)d_out;

  // ---- workspace partition ----
  char* wp = (char*)d_ws;
  auto alloc = [&](size_t bytes) -> void* {
    void* p = (void*)wp;
    wp += (bytes + 255) & ~(size_t)255;
    return p;
  };
  const size_t RQ = (size_t)BB * QQ;        // 4096 rows (q-side)
  const size_t RK = (size_t)BB * KVV;       // 8192 rows (kv-side)

  int* ids_draft = (int*)alloc(RQ * 4);
  int* ids_lbl   = (int*)alloc(RQ * 4);
  int* ids_pos   = (int*)alloc(RQ * 4);
  unsigned short* hcat_bf = (unsigned short*)alloc((size_t)BB * SS * (LL * DD) * 2);
  unsigned short* fc_bf   = (unsigned short*)alloc((size_t)DD * (LL * DD) * 2);
  unsigned short* wq_bf   = (unsigned short*)alloc((size_t)DD * DD * 2);
  unsigned short* wk_bf   = (unsigned short*)alloc((size_t)DD * DD * 2);
  unsigned short* wv_bf   = (unsigned short*)alloc((size_t)DD * DD * 2);
  unsigned short* wo_bf   = (unsigned short*)alloc((size_t)DD * DD * 2);
  unsigned short* lm_bf   = (unsigned short*)alloc((size_t)VV * DD * 2);
  unsigned short* ctx_bf  = (unsigned short*)alloc((size_t)BB * SS * DD * 2);
  unsigned short* emb_bf  = (unsigned short*)alloc(RQ * DD * 2);
  float*          emb_f   = (float*)alloc(RQ * DD * 4);
  float*          q_f     = (float*)alloc(RQ * DD * 4);
  float*          k_f     = (float*)alloc(RK * DD * 4);
  float*          v_f     = (float*)alloc(RK * DD * 4);
  unsigned short* q_bf    = (unsigned short*)alloc(RQ * DD * 2);
  unsigned short* k_bf    = (unsigned short*)alloc(RK * DD * 2);
  unsigned short* vT_bf   = (unsigned short*)alloc(RK * DD * 2);
  float*          ao_f    = (float*)alloc(RQ * DD * 4);
  unsigned short* ao_bf   = (unsigned short*)alloc(RQ * DD * 2);
  float*          proj_f  = (float*)alloc(RQ * DD * 4);
  unsigned short* h_bf    = (unsigned short*)alloc(RQ * DD * 2);
  float*          accum   = (float*)alloc(4 * 4);

  const int TB = 256;
  auto blocks = [](long n, int tb) { long b = (n + tb - 1) / tb; return (unsigned)(b > 16384 ? 16384 : b); };

  k_zero4<<<1, 32, 0, stream>>>(accum);
  k_build_ids<<<(unsigned)((RQ + TB - 1) / TB), TB, 0, stream>>>(input_ids, anchors, ids_draft, ids_lbl, ids_pos);

  long n_hcat = (long)BB * SS * LL * DD;
  k_hcat<<<blocks(n_hcat, TB), TB, 0, stream>>>(hidden, hcat_bf, n_hcat);
  k_cvt_bf16<<<blocks((long)DD * LL * DD, TB), TB, 0, stream>>>(fc_w, fc_bf, (long)DD * LL * DD);
  k_cvt_bf16<<<blocks((long)DD * DD, TB), TB, 0, stream>>>(wq, wq_bf, (long)DD * DD);
  k_cvt_bf16<<<blocks((long)DD * DD, TB), TB, 0, stream>>>(wk, wk_bf, (long)DD * DD);
  k_cvt_bf16<<<blocks((long)DD * DD, TB), TB, 0, stream>>>(wv, wv_bf, (long)DD * DD);
  k_cvt_bf16<<<blocks((long)DD * DD, TB), TB, 0, stream>>>(wo, wo_bf, (long)DD * DD);
  k_cvt_bf16<<<blocks((long)VV * DD, TB), TB, 0, stream>>>(lm_w, lm_bf, (long)VV * DD);
  k_emb_gather<<<blocks((long)RQ * DD, TB), TB, 0, stream>>>(embed, ids_draft, emb_bf, emb_f, (long)RQ * DD);

  // ctx = hcat @ fcW^T   (M=4096, N=1024, K=3072) -> bf16
  k_gemm<<<dim3(DD / 64, (unsigned)(BB * SS / 128)), TB, 0, stream>>>(
      hcat_bf, fc_bf, ctx_bf, BB * SS, DD, LL * DD, LL * DD, LL * DD, DD, 1);
  // q = emb @ wq^T -> f32
  k_gemm<<<dim3(DD / 64, (unsigned)(RQ / 128)), TB, 0, stream>>>(
      emb_bf, wq_bf, q_f, (int)RQ, DD, DD, DD, DD, DD, 0);
  // k,v: ctx part + draft part, per batch (to land in concatenated [B,KV,D] layout)
  for (int b = 0; b < BB; ++b) {
    k_gemm<<<dim3(DD / 64, SS / 128), TB, 0, stream>>>(
        ctx_bf + (size_t)b * SS * DD, wk_bf, k_f + (size_t)b * KVV * DD, SS, DD, DD, DD, DD, DD, 0);
    k_gemm<<<dim3(DD / 64, QQ / 128), TB, 0, stream>>>(
        emb_bf + (size_t)b * QQ * DD, wk_bf, k_f + ((size_t)b * KVV + SS) * DD, QQ, DD, DD, DD, DD, DD, 0);
    k_gemm<<<dim3(DD / 64, SS / 128), TB, 0, stream>>>(
        ctx_bf + (size_t)b * SS * DD, wv_bf, v_f + (size_t)b * KVV * DD, SS, DD, DD, DD, DD, DD, 0);
    k_gemm<<<dim3(DD / 64, QQ / 128), TB, 0, stream>>>(
        emb_bf + (size_t)b * QQ * DD, wv_bf, v_f + ((size_t)b * KVV + SS) * DD, QQ, DD, DD, DD, DD, DD, 0);
  }

  k_rope<<<blocks((long)RQ * HH * 64, TB), TB, 0, stream>>>(q_f, ids_pos, q_bf, (long)RQ * HH * 64, 0);
  k_rope<<<blocks((long)RK * HH * 64, TB), TB, 0, stream>>>(k_f, ids_pos, k_bf, (long)RK * HH * 64, 1);
  k_vtrans<<<blocks((long)RK * DD, TB), TB, 0, stream>>>(v_f, vT_bf, (long)RK * DD);

  k_attn<<<dim3(QQ / 128, HH, BB), TB, 0, stream>>>(q_bf, k_bf, vT_bf, anchors, ao_f);

  k_cvt_bf16<<<blocks((long)RQ * DD, TB), TB, 0, stream>>>(ao_f, ao_bf, (long)RQ * DD);
  k_gemm<<<dim3(DD / 64, (unsigned)(RQ / 128)), TB, 0, stream>>>(
      ao_bf, wo_bf, proj_f, (int)RQ, DD, DD, DD, DD, DD, 0);

  k_rmsnorm<<<(unsigned)RQ, TB, 0, stream>>>(emb_f, proj_f, norm_w, h_bf);

  k_lmloss<<<(unsigned)(RQ / 16), TB, 0, stream>>>(h_bf, lm_bf, ids_lbl, accum);
  k_final<<<1, 32, 0, stream>>>(accum, out);
  (void)in_sizes; (void)n_in; (void)out_size; (void)ws_size;
}
